// GCN_BAShapes_3513283248664
// MI455X (gfx1250) — matrix operand order repaired and checked by reading, not compiled, not run on hardware
//
#include <hip/hip_runtime.h>
#include <hip/hip_bf16.h>

typedef __attribute__((ext_vector_type(16))) _Float16 v16h;
typedef __attribute__((ext_vector_type(8)))  float    v8f;

#define T256 256

// ---------------- small utility kernels ----------------

__global__ __launch_bounds__(T256) void k_fill_f32(float* p, int n, float v) {
    int i = blockIdx.x * blockDim.x + threadIdx.x;
    if (i < n) p[i] = v;
}

__global__ __launch_bounds__(T256) void k_deg_count(const int* __restrict__ row,
                                                    float* __restrict__ deg, int E) {
    int e = blockIdx.x * blockDim.x + threadIdx.x;
    if (e < E) unsafeAtomicAdd(&deg[row[e]], 1.0f);
}

__global__ __launch_bounds__(T256) void k_rsqrt_inplace(float* p, int n) {
    int i = blockIdx.x * blockDim.x + threadIdx.x;
    if (i < n) p[i] = rsqrtf(p[i]);   // deg >= 1 always (self loops)
}

__global__ __launch_bounds__(T256) void k_f32_to_f16(const float* __restrict__ s,
                                                     _Float16* __restrict__ d, int n) {
    int i = blockIdx.x * blockDim.x + threadIdx.x;
    if (i < n) d[i] = (_Float16)s[i];
}

// ---------------- pack B (f32 K x NC row-major) into WMMA fragment layout ----------------
// pack[tile][lane][idx] with tile = (k/32)*tpr + (n/16); lane's 16 halves are
// b[s] = B[kb*32 + (lane>>4)*8 + (s&7) + (s>>3)*16][cb*16 + (lane&15)].
// GEMM then loads B as one contiguous, 32B-aligned v16h per lane per k-step.
__global__ __launch_bounds__(T256) void k_pack_b(const float* __restrict__ B,
                                                 _Float16* __restrict__ pack,
                                                 int K, int NC) {
    int pidx = blockIdx.x * blockDim.x + threadIdx.x;
    if (pidx >= K * NC) return;
    int tpr    = NC >> 4;
    int tile   = pidx >> 9;          // 512 halves per tile
    int within = pidx & 511;
    int lane   = within >> 4;
    int idx    = within & 15;
    int kb = tile / tpr, cb = tile % tpr;
    int k = kb * 32 + (lane >> 4) * 8 + (idx & 7) + (idx >> 3) * 16;     // table order
    int n = cb * 16 + (lane & 15);
    pack[pidx] = (_Float16)B[(size_t)k * NC + n];
}

// ---------------- WMMA GEMM: C[f32] = A[f16] x Bpack[f16] ----------------
// One wave computes one 16x16 output tile. K multiple of 32.
// Optionally fuses the GCN self-loop term: agg = C * dis[row]^2.
__global__ __launch_bounds__(T256) void k_gemm_wmma(const _Float16* __restrict__ A, int lda,
                                                    const _Float16* __restrict__ Bp,
                                                    float* __restrict__ C, int ldc,
                                                    float* __restrict__ aggOut,
                                                    const float* __restrict__ dis,
                                                    int nRows, int nCols, int K) {
    int wave = (blockIdx.x * blockDim.x + threadIdx.x) >> 5;
    int lane = threadIdx.x & 31;
    int tpr    = nCols >> 4;
    int nTiles = (nRows >> 4) * tpr;
    if (wave >= nTiles) return;                 // whole-wave exit: EXEC all-1s at WMMA
    int band = wave / tpr;
    int cb   = wave % tpr;
    int half = lane >> 4;
    int l16  = lane & 15;
    int rowA = band * 16 + l16;

    v8f acc = {};
    #pragma unroll 2
    for (int kk = 0; kk < K; kk += 32) {
        v16h a;
        // 16-bit A 16x32 layout: lanes 0-15 hold K=[0..7,16..23], lanes 16-31 K=[8..15,24..31]
        const _Float16* ap = A + (size_t)rowA * lda + kk + half * 8;
        #pragma unroll
        for (int i = 0; i < 8; ++i) { a[i] = ap[i]; a[8 + i] = ap[16 + i]; }
        // packed B: one contiguous 32B load per lane
        v16h b = *(const v16h*)(Bp + ((size_t)(kk >> 5) * tpr + cb) * 512 + lane * 16);
        acc = __builtin_amdgcn_wmma_f32_16x16x32_f16(false, a, false, b,
                                                     (short)0, acc, false, false);
    }
    // C/D layout: VGPR v -> row M = half*8 + v, col = l16
    int rowBase = band * 16 + half * 8;
    float* cp = C + (size_t)rowBase * ldc + cb * 16 + l16;
    #pragma unroll
    for (int v = 0; v < 8; ++v) cp[(size_t)v * ldc] = acc[v];
    if (aggOut) {
        float* gp = aggOut + (size_t)rowBase * ldc + cb * 16 + l16;
        #pragma unroll
        for (int v = 0; v < 8; ++v) {
            float d = dis[rowBase + v];
            gp[(size_t)v * ldc] = acc[v] * d * d;
        }
    }
}

// ---------------- edge scatter: agg[row,:] += h[col,:] * dis[row]*dis[col] ----------------
// 16 threads per edge, float4 each -> coalesced 256B gather per edge.
// Edge metadata loaded once per half-wave (lanes 0/16) and shuffled out.
__global__ __launch_bounds__(T256) void k_edge_scatter(const int* __restrict__ row,
                                                       const int* __restrict__ col,
                                                       const float* __restrict__ dis,
                                                       const float* __restrict__ h,
                                                       float* __restrict__ agg, int E) {
    int gid  = blockIdx.x * blockDim.x + threadIdx.x;
    int e    = gid >> 4;
    if (e >= E) return;
    int lane = threadIdx.x & 31;
    int part = lane & 15;
    int r = 0, c = 0;
    float nrm = 0.0f;
    if (part == 0) {                 // lanes 0 and 16 fetch for their half's edge
        r = row[e];
        c = col[e];
        nrm = dis[r] * dis[c];
    }
    int src = lane & 16;             // broadcast source lane per half
    r   = __shfl(r, src, 32);
    c   = __shfl(c, src, 32);
    nrm = __shfl(nrm, src, 32);

    float4 v = ((const float4*)h)[(size_t)c * 16 + part];
    float* dst = agg + (size_t)r * 64 + part * 4;
    unsafeAtomicAdd(dst + 0, v.x * nrm);
    unsafeAtomicAdd(dst + 1, v.y * nrm);
    unsafeAtomicAdd(dst + 2, v.z * nrm);
    unsafeAtomicAdd(dst + 3, v.w * nrm);
}

// ---------------- bias + relu + L2 norm; emit f16 slice of emb ----------------
// One wave32 per node; lane handles columns 2*lane, 2*lane+1.
__global__ __launch_bounds__(T256) void k_postproc(const float* __restrict__ agg,
                                                   const float* __restrict__ bias,
                                                   _Float16* __restrict__ emb, int embStride,
                                                   int colOff, int nNodes) {
    int node = (blockIdx.x * blockDim.x + threadIdx.x) >> 5;
    int lane = threadIdx.x & 31;
    if (node >= nNodes) return;
    float2 av = ((const float2*)(agg + (size_t)node * 64))[lane];
    float v0 = fmaxf(av.x + bias[2 * lane], 0.0f);
    float v1 = fmaxf(av.y + bias[2 * lane + 1], 0.0f);
    float s = v0 * v0 + v1 * v1;
    #pragma unroll
    for (int m = 16; m >= 1; m >>= 1) s += __shfl_xor(s, m, 32);
    float scale = 1.0f / fmaxf(sqrtf(s), 1e-12f);
    _Float16* d = emb + (size_t)node * embStride + colOff + 2 * lane;
    d[0] = (_Float16)(v0 * scale);
    d[1] = (_Float16)(v1 * scale);
}

// ---------------- final: WMMA emb[N x 192] @ Wlin[192 x 16] + b, log_softmax ----------------
__global__ __launch_bounds__(T256) void k_final_logsoftmax(const _Float16* __restrict__ A, int lda,
                                                           const _Float16* __restrict__ Bp, // packed
                                                           const float* __restrict__ bias, // [16]
                                                           float* __restrict__ out,
                                                           int nNodes, int K) {
    int wave = (blockIdx.x * blockDim.x + threadIdx.x) >> 5;
    int lane = threadIdx.x & 31;
    int nTiles = nNodes >> 4;
    if (wave >= nTiles) return;
    int half = lane >> 4;
    int l16  = lane & 15;
    int rowA = wave * 16 + l16;

    v8f acc = {};
    #pragma unroll 2
    for (int kk = 0; kk < K; kk += 32) {
        v16h a;
        const _Float16* ap = A + (size_t)rowA * lda + kk + half * 8;
        #pragma unroll
        for (int i = 0; i < 8; ++i) { a[i] = ap[i]; a[8 + i] = ap[16 + i]; }
        v16h b = *(const v16h*)(Bp + (size_t)(kk >> 5) * 512 + lane * 16);  // tpr == 1
        acc = __builtin_amdgcn_wmma_f32_16x16x32_f16(false, a, false, b,
                                                     (short)0, acc, false, false);
    }
    float bcol = bias[l16];
    // Each row (M = half*8 + v) lives across one 16-lane half at fragment index v.
    #pragma unroll
    for (int v = 0; v < 8; ++v) {
        float val = acc[v] + bcol;
        float mx = val;
        #pragma unroll
        for (int m = 8; m >= 1; m >>= 1) mx = fmaxf(mx, __shfl_xor(mx, m, 32));
        float ex = expf(val - mx);
        float sum = ex;
        #pragma unroll
        for (int m = 8; m >= 1; m >>= 1) sum += __shfl_xor(sum, m, 32);
        out[(size_t)(wave * 16 + half * 8 + v) * 16 + l16] = (val - mx) - logf(sum);
    }
}

// ---------------- host launcher ----------------

extern "C" void kernel_launch(void* const* d_in, const int* in_sizes, int n_in,
                              void* d_out, int out_size, void* d_ws, size_t ws_size,
                              hipStream_t stream) {
    const float* x    = (const float*)d_in[0];
    const int*   ei   = (const int*)d_in[1];
    const float* W1   = (const float*)d_in[2];
    const float* b1   = (const float*)d_in[3];
    const float* W2   = (const float*)d_in[4];
    const float* b2   = (const float*)d_in[5];
    const float* W3   = (const float*)d_in[6];
    const float* b3   = (const float*)d_in[7];
    const float* Wlin = (const float*)d_in[8];
    const float* blin = (const float*)d_in[9];

    const int N = in_sizes[0] / 64;    // 50000
    const int E = in_sizes[1] / 2;     // 1600000
    const int* rowIdx = ei;            // targets
    const int* colIdx = ei + E;        // sources

    // workspace carve-up (256B aligned)
    char* ws = (char*)d_ws;
    size_t off = 0;
    auto carve = [&](size_t bytes) -> void* {
        void* p = ws + off;
        off = (off + bytes + 255) & ~(size_t)255;
        return p;
    };
    float*    dis   = (float*)   carve((size_t)N * 4);
    float*    h     = (float*)   carve((size_t)N * 64 * 4);
    float*    agg   = (float*)   carve((size_t)N * 64 * 4);
    _Float16* emb16 = (_Float16*)carve((size_t)N * 192 * 2);
    _Float16* x16   = (_Float16*)carve((size_t)N * 64 * 2);
    _Float16* w16   = (_Float16*)carve((size_t)(3 * 64 * 64 + 192 * 16) * 2);

    auto blocks = [](long long n) { return (unsigned)((n + T256 - 1) / T256); };

    // degrees -> dis = rsqrt(deg), deg init 1.0 accounts for self loop
    k_fill_f32<<<blocks(N), T256, 0, stream>>>(dis, N, 1.0f);
    k_deg_count<<<blocks(E), T256, 0, stream>>>(rowIdx, dis, E);
    k_rsqrt_inplace<<<blocks(N), T256, 0, stream>>>(dis, N);

    // pack weights into WMMA B-fragment layout (also does f32->f16)
    k_pack_b<<<blocks(64 * 64), T256, 0, stream>>>(W1, w16 + 0 * 4096, 64, 64);
    k_pack_b<<<blocks(64 * 64), T256, 0, stream>>>(W2, w16 + 1 * 4096, 64, 64);
    k_pack_b<<<blocks(64 * 64), T256, 0, stream>>>(W3, w16 + 2 * 4096, 64, 64);
    k_pack_b<<<blocks(192 * 16), T256, 0, stream>>>(Wlin, w16 + 3 * 4096, 192, 16);
    // layer-0 input in f16
    k_f32_to_f16<<<blocks((long long)N * 64), T256, 0, stream>>>(x, x16, N * 64);

    const float* biases[3] = { b1, b2, b3 };
    const int gemmWaves  = (N / 16) * (64 / 16);               // one wave per 16x16 tile
    const unsigned gemmBlocks = (unsigned)((gemmWaves * 32 + T256 - 1) / T256);
    const unsigned postBlocks = (unsigned)(((long long)N * 32 + T256 - 1) / T256);
    const unsigned edgeBlocks = (unsigned)(((long long)E * 16 + T256 - 1) / T256);

    for (int l = 0; l < 3; ++l) {
        const _Float16* Ain = (l == 0) ? x16 : (emb16 + (l - 1) * 64);
        const int lda = (l == 0) ? 64 : 192;
        // GEMM fused with self-loop init: h = Ain @ Wl ; agg = h * dis^2
        k_gemm_wmma<<<gemmBlocks, T256, 0, stream>>>(Ain, lda, w16 + l * 4096,
                                                     h, 64, agg, dis, N, 64, 64);
        k_edge_scatter<<<edgeBlocks, T256, 0, stream>>>(rowIdx, colIdx, dis, h, agg, E);
        k_postproc<<<postBlocks, T256, 0, stream>>>(agg, biases[l], emb16, 192, l * 64, N);
    }

    const unsigned finBlocks = (unsigned)(((long long)(N / 16) * 32 + T256 - 1) / T256);
    k_final_logsoftmax<<<finBlocks, T256, 0, stream>>>(emb16, 192, w16 + 3 * 4096, blin,
                                                       (float*)d_out, N, 192);
}
